// Transformer_56178172231791
// MI455X (gfx1250) — matrix operand order repaired and checked by reading, not compiled, not run on hardware
//
#include <hip/hip_runtime.h>
#include <hip/hip_bf16.h>

// ---------------------------------------------------------------------------
// CDNA5 (gfx1250) transformer forward: 4 layers, talking-heads attention.
// All GEMMs via v_wmma_f32_16x16x32_f16 (f16 in, f32 accumulate).
// Wave tile: 32M x 64N (8 accumulators) for ~21 FLOP/B cache intensity.
// ---------------------------------------------------------------------------

typedef __attribute__((ext_vector_type(16))) _Float16 v16h;
typedef __attribute__((ext_vector_type(8)))  _Float16 v8h;
typedef __attribute__((ext_vector_type(8)))  float    v8f;

#define BD   8      // batch
#define SN   512    // sequence
#define DIMD 768
#define HD   12
#define DHD  64
#define INNERD 768
#define MLPD 3072
#define LSF  0.1f

// --------------------------- weight convert (transpose + f16) --------------
// src [R,C] f32 row-major -> dst [C,R] f16 row-major
__global__ __launch_bounds__(256)
void transpose_cast_kernel(const float* __restrict__ src, _Float16* __restrict__ dst,
                           int R, int C) {
    long long i = (long long)blockIdx.x * 256 + threadIdx.x;
    long long total = (long long)R * C;
    if (i >= total) return;
    int c = (int)(i / R);
    int r = (int)(i % R);
    dst[i] = (_Float16)src[(long long)r * C + c];
}

// v slice of kv [b,n,2*INNER] -> vt [b,h,dh,n] f16
__global__ __launch_bounds__(256)
void extract_vt_kernel(const _Float16* __restrict__ kv, _Float16* __restrict__ vt) {
    long long i = (long long)blockIdx.x * 256 + threadIdx.x;
    if (i >= (long long)BD * HD * DHD * SN) return;
    int n = (int)(i % SN);
    int d = (int)((i / SN) % DHD);
    int h = (int)((i / ((long long)SN * DHD)) % HD);
    int b = (int)(i / ((long long)SN * DHD * HD));
    vt[i] = kv[((long long)b * SN + n) * (2 * INNERD) + INNERD + h * DHD + d];
}

// --------------------------- LayerNorm (fp32 in -> f16 out) ----------------
__global__ __launch_bounds__(256)
void ln_kernel(const float* __restrict__ x, const float* __restrict__ gamma,
               _Float16* __restrict__ out) {
    __shared__ float red[256];
    int row = blockIdx.x;
    int tid = threadIdx.x;
    const float* xr = x + (long long)row * DIMD;
    float v0 = xr[tid], v1 = xr[tid + 256], v2 = xr[tid + 512];
    red[tid] = v0 + v1 + v2;
    __syncthreads();
    for (int s = 128; s > 0; s >>= 1) { if (tid < s) red[tid] += red[tid + s]; __syncthreads(); }
    float mean = red[0] * (1.0f / DIMD);
    __syncthreads();
    float d0 = v0 - mean, d1 = v1 - mean, d2 = v2 - mean;
    red[tid] = d0 * d0 + d1 * d1 + d2 * d2;
    __syncthreads();
    for (int s = 128; s > 0; s >>= 1) { if (tid < s) red[tid] += red[tid + s]; __syncthreads(); }
    float inv = rsqrtf(red[0] * (1.0f / DIMD) + 1e-5f);
    _Float16* orow = out + (long long)row * DIMD;
    orow[tid]       = (_Float16)(d0 * inv * gamma[tid]);
    orow[tid + 256] = (_Float16)(d1 * inv * gamma[tid + 256]);
    orow[tid + 512] = (_Float16)(d2 * inv * gamma[tid + 512]);
}

// --------------------------- fused mix -> softmax -> mix --------------------
// block = one (b,i) row; dots f32 [b,h,i,j] -> attn f16 [b,g,i,j]
__global__ __launch_bounds__(256)
void msm_kernel(const float* __restrict__ dots, const float* __restrict__ mp,
                const float* __restrict__ mq, _Float16* __restrict__ attn) {
    __shared__ float probs[HD][SN];   // 24 KB
    __shared__ float red[256];
    __shared__ float mpS[HD * HD], mqS[HD * HD];
    int tid = threadIdx.x;
    int bi  = blockIdx.x;
    int b = bi >> 9, i = bi & 511;
    if (tid < HD * HD) { mpS[tid] = mp[tid]; mqS[tid] = mq[tid]; }
    __syncthreads();
    const float* base = dots + ((long long)b * HD * SN + i) * SN;
    for (int jj = tid; jj < SN; jj += 256) {
        float d[HD];
        #pragma unroll
        for (int h = 0; h < HD; h++) d[h] = base[(long long)h * SN * SN + jj];
        #pragma unroll
        for (int g = 0; g < HD; g++) {
            float m = 0.f;
            #pragma unroll
            for (int h = 0; h < HD; h++) m += d[h] * mpS[h * HD + g];
            probs[g][jj] = m;
        }
    }
    __syncthreads();
    for (int g = 0; g < HD; g++) {
        float v0 = probs[g][tid], v1 = probs[g][tid + 256];
        red[tid] = fmaxf(v0, v1);
        __syncthreads();
        for (int s = 128; s > 0; s >>= 1) { if (tid < s) red[tid] = fmaxf(red[tid], red[tid + s]); __syncthreads(); }
        float mx = red[0];
        __syncthreads();
        float e0 = __expf(v0 - mx), e1 = __expf(v1 - mx);
        red[tid] = e0 + e1;
        __syncthreads();
        for (int s = 128; s > 0; s >>= 1) { if (tid < s) red[tid] += red[tid + s]; __syncthreads(); }
        float inv = 1.f / red[0];
        __syncthreads();
        probs[g][tid] = e0 * inv;
        probs[g][tid + 256] = e1 * inv;
    }
    __syncthreads();
    _Float16* abase = attn + ((long long)b * HD * SN + i) * SN;
    for (int jj = tid; jj < SN; jj += 256) {
        float p[HD];
        #pragma unroll
        for (int g = 0; g < HD; g++) p[g] = probs[g][jj];
        #pragma unroll
        for (int g2 = 0; g2 < HD; g2++) {
            float o = 0.f;
            #pragma unroll
            for (int g = 0; g < HD; g++) o += p[g] * mqS[g * HD + g2];
            abase[(long long)g2 * SN * SN + jj] = (_Float16)o;
        }
    }
}

// --------------------------- WMMA GEMM -------------------------------------
// C[M,N] = A[M,K](f16, lda) @ Bt[N,K]^T (f16, ldb), batched via z = zb*Hc+zh.
// Wave tile: 32M x 64N. Block = 8 waves.
// MODE 0: f16 store   MODE 1: f32 store * alpha
// MODE 2: gelu(acc+bias)->f16   MODE 3: (acc+bias)*LS + resid -> f32
template<int MODE>
__global__ __launch_bounds__(256)
void wmma_gemm_kernel(const _Float16* __restrict__ A, const _Float16* __restrict__ Bt,
                      void* __restrict__ C, const float* __restrict__ bias,
                      float* __restrict__ resid,
                      int M, int N, int K, int lda, int ldb, int ldc,
                      int Hc,
                      long long aB, long long aH, long long bB, long long bH,
                      long long cB, long long cH, float alpha) {
    int z  = blockIdx.y;
    int zb = z / Hc, zh = z % Hc;
    const _Float16* Ab = A  + (long long)zb * aB + (long long)zh * aH;
    const _Float16* Bb = Bt + (long long)zb * bB + (long long)zh * bH;
    long long cbase = (long long)zb * cB + (long long)zh * cH;

    int lane = threadIdx.x & 31;
    int wave = threadIdx.x >> 5;
    int tilesN = N >> 6;                           // wave tile: 32M x 64N
    int wt = blockIdx.x * 8 + wave;
    if (wt >= (M >> 5) * tilesN) return;
    int row0 = (wt / tilesN) << 5;
    int col0 = (wt % tilesN) << 6;

    int r0  = row0 + (lane & 15);
    int kbA = (lane >> 4) << 4;                    // 0 or 16
    int cb0 = col0 + (lane & 15);
    int kbB = (lane >> 4) << 4;                    // 0 or 16
    const _Float16* ap0 = Ab + (long long)r0 * lda + kbA;
    const _Float16* ap1 = ap0 + (long long)16 * lda;
    const _Float16* bp  = Bb + (long long)cb0 * ldb + kbB;

    v8f acc00 = {}, acc01 = {}, acc02 = {}, acc03 = {};
    v8f acc10 = {}, acc11 = {}, acc12 = {}, acc13 = {};
    for (int k0 = 0; k0 < K; k0 += 32) {
        v8h lo0 = *(const v8h*)(ap0 + k0);
        v8h hi0 = *(const v8h*)(ap0 + k0 + 8);
        v8h lo1 = *(const v8h*)(ap1 + k0);
        v8h hi1 = *(const v8h*)(ap1 + k0 + 8);
        v16h a0, a1;
        #pragma unroll
        for (int i = 0; i < 8; i++) {
            a0[i] = lo0[i]; a0[8 + i] = hi0[i];
            a1[i] = lo1[i]; a1[8 + i] = hi1[i];
        }
        v16h b0 = *(const v16h*)(bp + k0);
        v16h b1 = *(const v16h*)(bp + (long long)16 * ldb + k0);
        v16h b2 = *(const v16h*)(bp + (long long)32 * ldb + k0);
        v16h b3 = *(const v16h*)(bp + (long long)48 * ldb + k0);
        acc00 = __builtin_amdgcn_wmma_f32_16x16x32_f16(false, a0, false, b0, (short)0, acc00, false, false);
        acc10 = __builtin_amdgcn_wmma_f32_16x16x32_f16(false, a1, false, b0, (short)0, acc10, false, false);
        acc01 = __builtin_amdgcn_wmma_f32_16x16x32_f16(false, a0, false, b1, (short)0, acc01, false, false);
        acc11 = __builtin_amdgcn_wmma_f32_16x16x32_f16(false, a1, false, b1, (short)0, acc11, false, false);
        acc02 = __builtin_amdgcn_wmma_f32_16x16x32_f16(false, a0, false, b2, (short)0, acc02, false, false);
        acc12 = __builtin_amdgcn_wmma_f32_16x16x32_f16(false, a1, false, b2, (short)0, acc12, false, false);
        acc03 = __builtin_amdgcn_wmma_f32_16x16x32_f16(false, a0, false, b3, (short)0, acc03, false, false);
        acc13 = __builtin_amdgcn_wmma_f32_16x16x32_f16(false, a1, false, b3, (short)0, acc13, false, false);
    }

    int cn0 = lane & 15;
    int cm0 = row0 + ((lane >> 4) << 3);
    #pragma unroll
    for (int half = 0; half < 2; half++) {
        #pragma unroll
        for (int t = 0; t < 4; t++) {
            v8f accv;
            if (half == 0) accv = (t == 0) ? acc00 : (t == 1) ? acc01 : (t == 2) ? acc02 : acc03;
            else           accv = (t == 0) ? acc10 : (t == 1) ? acc11 : (t == 2) ? acc12 : acc13;
            int ncol = col0 + t * 16 + cn0;
            int mrow = cm0 + half * 16;
            #pragma unroll
            for (int rr = 0; rr < 8; rr++) {
                long long idx = cbase + (long long)(mrow + rr) * ldc + ncol;
                float av = accv[rr];
                if (MODE == 0) {
                    ((_Float16*)C)[idx] = (_Float16)av;
                } else if (MODE == 1) {
                    ((float*)C)[idx] = av * alpha;
                } else if (MODE == 2) {
                    float t2 = av + bias[ncol];
                    float g  = 0.5f * t2 * (1.f + tanhf(0.7978845608028654f * (t2 + 0.044715f * t2 * t2 * t2)));
                    ((_Float16*)C)[idx] = (_Float16)g;
                } else {
                    ((float*)C)[idx] = (av + bias[ncol]) * LSF + resid[idx];
                }
            }
        }
    }
}

// --------------------------- host-side orchestration ------------------------
static inline void gemm(int mode, const _Float16* A, const _Float16* B, void* C,
                        const float* bias, float* resid,
                        int M, int N, int K, int lda, int ldb, int ldc,
                        int batches, int Hc,
                        long long aB, long long aH, long long bB, long long bH,
                        long long cB, long long cH, float alpha, hipStream_t s) {
    int totalWT = (M >> 5) * (N >> 6);
    dim3 grid((totalWT + 7) / 8, batches);
    switch (mode) {
    case 0: wmma_gemm_kernel<0><<<grid, 256, 0, s>>>(A, B, C, bias, resid, M, N, K, lda, ldb, ldc, Hc, aB, aH, bB, bH, cB, cH, alpha); break;
    case 1: wmma_gemm_kernel<1><<<grid, 256, 0, s>>>(A, B, C, bias, resid, M, N, K, lda, ldb, ldc, Hc, aB, aH, bB, bH, cB, cH, alpha); break;
    case 2: wmma_gemm_kernel<2><<<grid, 256, 0, s>>>(A, B, C, bias, resid, M, N, K, lda, ldb, ldc, Hc, aB, aH, bB, bH, cB, cH, alpha); break;
    default: wmma_gemm_kernel<3><<<grid, 256, 0, s>>>(A, B, C, bias, resid, M, N, K, lda, ldb, ldc, Hc, aB, aH, bB, bH, cB, cH, alpha); break;
    }
}

extern "C" void kernel_launch(void* const* d_in, const int* in_sizes, int n_in,
                              void* d_out, int out_size, void* d_ws, size_t ws_size,
                              hipStream_t stream) {
    (void)in_sizes; (void)n_in; (void)out_size; (void)ws_size;

    const float* x_in = (const float*)d_in[0];
    const float* ln1  = (const float*)d_in[1];
    const float* Wq   = (const float*)d_in[2];
    const float* Wkv  = (const float*)d_in[3];
    const float* mp   = (const float*)d_in[4];
    const float* mq   = (const float*)d_in[5];
    const float* Wo   = (const float*)d_in[6];
    const float* bo   = (const float*)d_in[7];
    const float* ln2  = (const float*)d_in[8];
    const float* W1   = (const float*)d_in[9];
    const float* b1   = (const float*)d_in[10];
    const float* W2   = (const float*)d_in[11];
    const float* b2   = (const float*)d_in[12];

    const long long TOK = (long long)BD * SN;                 // 4096
    const long long XSZ = TOK * DIMD;                         // 3,145,728

    char* w = (char*)d_ws;
    float*    x_cur = (float*)(w + 0);                        // 12.58 MB
    _Float16* h16   = (_Float16*)(w + 12582912);              // 6.29 MB
    _Float16* q16   = (_Float16*)(w + 18874368);              // 6.29 MB
    _Float16* kv16  = (_Float16*)(w + 25165824);              // 12.58 MB
    _Float16* vt16  = (_Float16*)(w + 37748736);              // 6.29 MB
    _Float16* o16   = (_Float16*)(w + 44040192);              // 6.29 MB
    _Float16* mlp16 = (_Float16*)(w + 50331648);              // 25.17 MB
    float*    dots  = (float*)(w + 75497472);                 // 100.66 MB
    _Float16* att16 = (_Float16*)(w + 176160768);             // 50.33 MB
    _Float16* WqT   = (_Float16*)(w + 226492416);
    _Float16* WkvT  = (_Float16*)(w + 231211008);
    _Float16* WoT   = (_Float16*)(w + 240648192);
    _Float16* W1T   = (_Float16*)(w + 245366784);
    _Float16* W2T   = (_Float16*)(w + 264241152);

    // weights -> transposed f16, all layers
    for (int l = 0; l < 4; l++) {
        long long e;
        e = (long long)DIMD * INNERD;
        transpose_cast_kernel<<<(int)((e + 255) / 256), 256, 0, stream>>>(Wq + l * e, WqT + l * e, DIMD, INNERD);
        e = (long long)DIMD * 2 * INNERD;
        transpose_cast_kernel<<<(int)((e + 255) / 256), 256, 0, stream>>>(Wkv + l * e, WkvT + l * e, DIMD, 2 * INNERD);
        e = (long long)INNERD * DIMD;
        transpose_cast_kernel<<<(int)((e + 255) / 256), 256, 0, stream>>>(Wo + l * e, WoT + l * e, INNERD, DIMD);
        e = (long long)DIMD * MLPD;
        transpose_cast_kernel<<<(int)((e + 255) / 256), 256, 0, stream>>>(W1 + l * e, W1T + l * e, DIMD, MLPD);
        e = (long long)MLPD * DIMD;
        transpose_cast_kernel<<<(int)((e + 255) / 256), 256, 0, stream>>>(W2 + l * e, W2T + l * e, MLPD, DIMD);
    }

    hipMemcpyAsync(x_cur, x_in, XSZ * sizeof(float), hipMemcpyDeviceToDevice, stream);

    const long long NN = (long long)SN * SN;
    for (int l = 0; l < 4; l++) {
        // ---- attention ----
        ln_kernel<<<(int)TOK, 256, 0, stream>>>(x_cur, ln1 + l * DIMD, h16);
        gemm(0, h16, WqT + (long long)l * DIMD * INNERD, q16, nullptr, nullptr,
             (int)TOK, INNERD, DIMD, DIMD, DIMD, INNERD,
             1, 1, 0, 0, 0, 0, 0, 0, 1.f, stream);
        gemm(0, h16, WkvT + (long long)l * DIMD * 2 * INNERD, kv16, nullptr, nullptr,
             (int)TOK, 2 * INNERD, DIMD, DIMD, DIMD, 2 * INNERD,
             1, 1, 0, 0, 0, 0, 0, 0, 1.f, stream);
        // scores: dots[b,h,i,j] = (q . k) * dh^-0.5   (batched over b*h)
        gemm(1, q16, kv16, dots, nullptr, nullptr,
             SN, SN, DHD, INNERD, 2 * INNERD, SN,
             BD * HD, HD,
             (long long)SN * INNERD, DHD,
             (long long)SN * 2 * INNERD, DHD,
             (long long)HD * NN, NN, 0.125f, stream);
        msm_kernel<<<(int)TOK, 256, 0, stream>>>(dots, mp + l * HD * HD, mq + l * HD * HD, att16);
        extract_vt_kernel<<<(int)(((long long)BD * HD * DHD * SN + 255) / 256), 256, 0, stream>>>(kv16, vt16);
        // o[b,n,h*d] = attn @ v   (batched over b*h)
        gemm(0, att16, vt16, o16, nullptr, nullptr,
             SN, DHD, SN, SN, SN, INNERD,
             BD * HD, HD,
             (long long)HD * NN, NN,
             (long long)HD * DHD * SN, (long long)DHD * SN,
             (long long)SN * INNERD, DHD, 1.f, stream);
        // x = (o @ Wo + bo) * LS + x
        gemm(3, o16, WoT + (long long)l * INNERD * DIMD, x_cur, bo + l * DIMD, x_cur,
             (int)TOK, DIMD, INNERD, INNERD, INNERD, DIMD,
             1, 1, 0, 0, 0, 0, 0, 0, 1.f, stream);
        // ---- MLP ----
        ln_kernel<<<(int)TOK, 256, 0, stream>>>(x_cur, ln2 + l * DIMD, h16);
        gemm(2, h16, W1T + (long long)l * DIMD * MLPD, mlp16, b1 + l * MLPD, nullptr,
             (int)TOK, MLPD, DIMD, DIMD, DIMD, MLPD,
             1, 1, 0, 0, 0, 0, 0, 0, 1.f, stream);
        gemm(3, mlp16, W2T + (long long)l * MLPD * DIMD, x_cur, b2 + l * DIMD, x_cur,
             (int)TOK, DIMD, MLPD, MLPD, MLPD, DIMD,
             1, 1, 0, 0, 0, 0, 0, 0, 1.f, stream);
    }

    hipMemcpyAsync(d_out, x_cur, XSZ * sizeof(float), hipMemcpyDeviceToDevice, stream);
}
